// GaussianUpsamplingModule_33913061769671
// MI455X (gfx1250) — matrix operand order repaired and checked by reading, not compiled, not run on hardware
//
#include <hip/hip_runtime.h>
#include <hip/hip_fp16.h>

// ---------------------------------------------------------------------------
// Gaussian upsampling (FastSpeech-style) for MI455X / gfx1250 (wave32, WMMA).
//
// Pipeline:
//   k_xc_ranges  : fused 3x conv1d(K=3) + xc build (f16) + proj/softplus -> stds
//   k_means      : per-batch cumsum of integer durations -> means
//   k_denom      : denom[b,t] = sum_l probs[b,l,t]
//   k_weights_t  : weights f32 -> d_out (coalesced) AND f16
//                  A matrix As (B,Tp,L) via in-kernel 32x32 LDS transpose
//   k_transpose  : xcH (L,E) -> xcT (E,L) so B fragments are contiguous
//   k_gemm       : x_upsamp[b] = W^T (T x L) @ xc (L x E)
//                  via v_wmma_f32_16x16x32_f16, 16x64 C strip per wave,
//                  software-pipelined fragment loads.
// ---------------------------------------------------------------------------

typedef _Float16 f16;
typedef __attribute__((ext_vector_type(16))) _Float16 v16h;
typedef __attribute__((ext_vector_type(8)))  float    v8f;

#define BB 16
#define LL 512
#define EE 512

__device__ __constant__ float kLogSqrt2Pi = 0.9189385332046727f;

// K-swizzle for the A operand: within each 32-wide K block, swap the two
// middle 8-groups so the per-lane fragment (ISA 16-bit A 16x32 layout:
// h=0 -> K {kk..kk+7, kk+16..kk+23}, h=1 -> K {kk+8..15, kk+24..31})
// becomes a single contiguous 16-half read at offset kk + h*16.
__device__ __forceinline__ int kswz(int k) {
    int sub = k & 31;
    int g   = sub >> 3;
    g = (g == 1) ? 2 : (g == 2) ? 1 : g;
    return (k & ~31) | (g << 3) | (sub & 7);
}

// ---------------------------------------------------------------------------
// Kernel 1: per (b,l) row. 256 threads over E=512.
// ---------------------------------------------------------------------------
__global__ void k_xc_ranges(const float* __restrict__ x,
                            const float* __restrict__ durf,
                            const float* __restrict__ energies,
                            const float* __restrict__ pitch,
                            const int*   __restrict__ lens,
                            const float* __restrict__ w_dur, const float* __restrict__ b_dur,
                            const float* __restrict__ w_nrg, const float* __restrict__ b_nrg,
                            const float* __restrict__ w_pit, const float* __restrict__ b_pit,
                            const float* __restrict__ w_proj, const float* __restrict__ b_proj,
                            f16*   __restrict__ xcH,      // (B, L, E) f16
                            float* __restrict__ stds,     // (B, L)
                            float* __restrict__ logstds)  // (B, L)
{
    const int row = blockIdx.x;            // b*L + l
    const int b = row >> 9;
    const int l = row & (LL - 1);

    const float dm = (l > 0)      ? durf[row - 1]     : 0.f;
    const float d0 = durf[row];
    const float dp = (l < LL - 1) ? durf[row + 1]     : 0.f;
    const float em = (l > 0)      ? energies[row - 1] : 0.f;
    const float e0 = energies[row];
    const float ep = (l < LL - 1) ? energies[row + 1] : 0.f;
    const float pm = (l > 0)      ? pitch[row - 1]    : 0.f;
    const float p0 = pitch[row];
    const float pp = (l < LL - 1) ? pitch[row + 1]    : 0.f;

    float acc = 0.f;
    for (int e = threadIdx.x; e < EE; e += blockDim.x) {
        const float nrg = w_nrg[3 * e] * em + w_nrg[3 * e + 1] * e0 + w_nrg[3 * e + 2] * ep + b_nrg[e];
        const float pit = w_pit[3 * e] * pm + w_pit[3 * e + 1] * p0 + w_pit[3 * e + 2] * pp + b_pit[e];
        const float dur = w_dur[3 * e] * dm + w_dur[3 * e + 1] * d0 + w_dur[3 * e + 2] * dp + b_dur[e];
        const float xc  = x[(size_t)row * EE + e] + nrg + pit;
        xcH[(size_t)row * EE + e] = (f16)xc;
        acc += (xc + dur) * w_proj[e];
    }

    __shared__ float red[256];
    red[threadIdx.x] = acc;
    __syncthreads();
    for (int s = 128; s > 0; s >>= 1) {
        if ((int)threadIdx.x < s) red[threadIdx.x] += red[threadIdx.x + s];
        __syncthreads();
    }
    if (threadIdx.x == 0) {
        const float r   = red[0] + b_proj[0];
        const float sp  = (r > 20.f) ? r : log1pf(expf(r));   // stable softplus
        const bool  pad = (l >= lens[b]);
        const float v   = pad ? 1.f : sp;
        const float sd  = fmaxf(v, 0.001f);                   // clip min 0.001
        stds[row]    = sd;
        logstds[row] = logf(sd);
    }
}

// ---------------------------------------------------------------------------
// Kernel 2: means[b,l] = csum(d) - 0.5*d
// ---------------------------------------------------------------------------
__global__ void k_means(const int* __restrict__ duri, float* __restrict__ means)
{
    const int b = blockIdx.x;
    if (threadIdx.x == 0) {
        float csum = 0.f;
        for (int l = 0; l < LL; ++l) {
            const float d = (float)duri[b * LL + l];
            csum += d;
            means[b * LL + l] = csum - 0.5f * d;
        }
    }
}

// ---------------------------------------------------------------------------
// Kernel 3a: denom[b,t] = sum_l probs + 1e-20
// ---------------------------------------------------------------------------
__global__ void k_denom(const float* __restrict__ frames,
                        const float* __restrict__ means,
                        const float* __restrict__ stds,
                        const float* __restrict__ logstds,
                        const int*   __restrict__ lens,
                        int T,
                        float* __restrict__ denom)
{
    const int idx = blockIdx.x;          // b*T + t
    const int b = idx / T;
    const int t = idx - b * T;
    const float ft  = frames[t];
    const int   len = lens[b];

    float acc = 0.f;
    for (int l = threadIdx.x; l < LL; l += blockDim.x) {
        if (l < len) {
            const int r = b * LL + l;
            const float z = (ft - means[r]) / stds[r];
            acc += expf(-0.5f * z * z - logstds[r] - kLogSqrt2Pi);
        }
    }
    __shared__ float red[256];
    red[threadIdx.x] = acc;
    __syncthreads();
    for (int s = 128; s > 0; s >>= 1) {
        if ((int)threadIdx.x < s) red[threadIdx.x] += red[threadIdx.x + s];
        __syncthreads();
    }
    if (threadIdx.x == 0) denom[idx] = red[0] + 1e-20f;
}

// ---------------------------------------------------------------------------
// Kernel 3b (fused): 32(l) x 32(t) tile per block.
//   - weights f32 -> d_out (B,L,T), coalesced in t
//   - f16 weights staged in LDS, written transposed to
//     As (B, Tp, L): As[b][t][l] -- GEMM A operand, zero extra traffic.
// ---------------------------------------------------------------------------
__global__ void k_weights_t(const float* __restrict__ frames,
                            const float* __restrict__ means,
                            const float* __restrict__ stds,
                            const float* __restrict__ logstds,
                            const float* __restrict__ denom,
                            const int*   __restrict__ lens,
                            int T, int Tp,
                            float* __restrict__ wOut,   // (B, L, T) f32
                            f16*   __restrict__ As)     // (B, Tp, L) f16
{
    __shared__ f16 tile[32][33];
    const int b  = blockIdx.z;
    const int l0 = blockIdx.y * 32;
    const int t0 = blockIdx.x * 32;
    const int tx = threadIdx.x & 31;     // t offset
    const int ty = threadIdx.x >> 5;     // 0..7 (l offset, step 8)

    const int   t   = t0 + tx;
    const int   len = lens[b];
    const bool  tin = (t < T);
    const float ft  = tin ? frames[t]       : 0.f;
    const float dn  = tin ? denom[b * T + t] : 1.f;

#pragma unroll
    for (int i = 0; i < 32; i += 8) {
        const int l   = l0 + ty + i;
        const int row = b * LL + l;
        float w = 0.f;
        if (tin && l < len) {
            const float z = (ft - means[row]) / stds[row];
            w = expf(-0.5f * z * z - logstds[row] - kLogSqrt2Pi) / dn;
        }
        if (tin) wOut[(size_t)row * T + t] = w;
        tile[ty + i][tx] = (f16)w;       // tile[l - l0][t - t0]
    }
    __syncthreads();
#pragma unroll
    for (int i = 0; i < 32; i += 8) {
        const int t2 = t0 + ty + i;      // output row (t)
        const int l2 = l0 + tx;          // output col (K dim = l)
        if (t2 < T)
            As[((size_t)b * Tp + t2) * LL + l2] = tile[tx][ty + i];
    }
}

// ---------------------------------------------------------------------------
// Kernel T: batched f16 transpose xcH (L,E) -> xcT (E,L), 32x32 LDS tiles.
// ---------------------------------------------------------------------------
__global__ void k_transpose_f16(const f16* __restrict__ in, f16* __restrict__ out)
{
    __shared__ f16 tile[32][33];
    const int b  = blockIdx.z;
    const int r0 = blockIdx.y * 32;      // input row tile (l)
    const int c0 = blockIdx.x * 32;      // input col tile (e)
    const f16* I = in  + (size_t)b * LL * EE;
    f16*       O = out + (size_t)b * EE * LL;
    const int tx = threadIdx.x & 31;
    const int ty = threadIdx.x >> 5;     // 0..7

#pragma unroll
    for (int i = 0; i < 32; i += 8)
        tile[ty + i][tx] = I[(size_t)(r0 + ty + i) * EE + c0 + tx];
    __syncthreads();
#pragma unroll
    for (int i = 0; i < 32; i += 8)
        O[(size_t)(c0 + ty + i) * LL + r0 + tx] = tile[tx][ty + i];
}

// ---------------------------------------------------------------------------
// Kernel 4: batched GEMM  C(T x E) = A(T x L) * X(L x E), f16 in / f32 accum.
// One wave per 16(M) x 64(N) C strip: 1 A load + 4 B loads + 4 WMMA per
// 32-K step; K = L = 512 -> 64 x v_wmma_f32_16x16x32_f16 per wave.
// Software-pipelined: iteration k+1's five fragments are loaded before
// iteration k's WMMAs so the load clauses hide behind the matrix ops.
// All fragments are single contiguous 32B per-lane loads:
//   A: As[b][mBase+r][kk + h*16 .. +15]   (k = kk + h*16 + s, same order as B)
//   B: xcT[b][nBase+q*16+r][kk + h*16 .. +15]
// C (f32 16x16): lane n = lane&15, vgpr v -> m = h*8 + v.
// ---------------------------------------------------------------------------
__global__ void k_gemm(const f16* __restrict__ As,   // (B, Tp, L)
                       const f16* __restrict__ xcT,  // (B, E, L)
                       float* __restrict__ out,      // (B, T, E)
                       int T, int Tp, int Mtiles)
{
    const int wave = (int)((blockIdx.x * blockDim.x + threadIdx.x) >> 5);
    const int lane = threadIdx.x & 31;
    const int nGroups = EE / 64;                      // 8
    const int tilesPerBatch = Mtiles * nGroups;
    if (wave >= BB * tilesPerBatch) return;           // wave-uniform

    const int b   = wave / tilesPerBatch;
    const int rem = wave - b * tilesPerBatch;
    const int mt  = rem / nGroups;
    const int ng  = rem - mt * nGroups;
    const int mBase = mt * 16;
    const int nBase = ng * 64;

    const int r = lane & 15;
    const int h = lane >> 4;

    const f16* Arow = As  + ((size_t)b * Tp + mBase + r) * LL + h * 16;
    const f16* X0   = xcT + ((size_t)b * EE + nBase + r) * LL + h * 16;

    v8f acc[4] = {v8f{}, v8f{}, v8f{}, v8f{}};

    // prologue: fragments for kk = 0
    v16h a  = *(const v16h*)(Arow);
    v16h b0 = *(const v16h*)(X0);
    v16h b1 = *(const v16h*)(X0 + (size_t)16 * LL);
    v16h b2 = *(const v16h*)(X0 + (size_t)32 * LL);
    v16h b3 = *(const v16h*)(X0 + (size_t)48 * LL);

#pragma unroll
    for (int kk = 0; kk < LL; kk += 32) {
        v16h an = a, bn0 = b0, bn1 = b1, bn2 = b2, bn3 = b3;
        if (kk + 32 < LL) {               // compile-time after unroll
            const int kn = kk + 32;
            an  = *(const v16h*)(Arow + kn);
            bn0 = *(const v16h*)(X0 + kn);
            bn1 = *(const v16h*)(X0 + (size_t)16 * LL + kn);
            bn2 = *(const v16h*)(X0 + (size_t)32 * LL + kn);
            bn3 = *(const v16h*)(X0 + (size_t)48 * LL + kn);
        }
        acc[0] = __builtin_amdgcn_wmma_f32_16x16x32_f16(false, a, false, b0, (short)0, acc[0], false, false);
        acc[1] = __builtin_amdgcn_wmma_f32_16x16x32_f16(false, a, false, b1, (short)0, acc[1], false, false);
        acc[2] = __builtin_amdgcn_wmma_f32_16x16x32_f16(false, a, false, b2, (short)0, acc[2], false, false);
        acc[3] = __builtin_amdgcn_wmma_f32_16x16x32_f16(false, a, false, b3, (short)0, acc[3], false, false);
        a = an; b0 = bn0; b1 = bn1; b2 = bn2; b3 = bn3;
    }

    float* C = out + (size_t)b * T * EE;
    if (mBase + 16 <= T) {
        // full tile: unguarded stores
#pragma unroll
        for (int q = 0; q < 4; ++q)
#pragma unroll
            for (int v = 0; v < 8; ++v)
                C[(size_t)(mBase + h * 8 + v) * EE + nBase + q * 16 + r] = acc[q][v];
    } else {
        // tail tile
#pragma unroll
        for (int q = 0; q < 4; ++q)
#pragma unroll
            for (int v = 0; v < 8; ++v) {
                const int m = mBase + h * 8 + v;
                if (m < T)
                    C[(size_t)m * EE + nBase + q * 16 + r] = acc[q][v];
            }
    }
}

// ---------------------------------------------------------------------------
extern "C" void kernel_launch(void* const* d_in, const int* in_sizes, int n_in,
                              void* d_out, int out_size, void* d_ws, size_t ws_size,
                              hipStream_t stream)
{
    const float* x        = (const float*)d_in[0];
    const float* durf     = (const float*)d_in[1];
    const int*   duri     = (const int*)  d_in[2];
    const float* energies = (const float*)d_in[3];
    const float* pitch    = (const float*)d_in[4];
    const int*   lens     = (const int*)  d_in[5];
    const float* frames   = (const float*)d_in[6];
    const float* w_dur    = (const float*)d_in[7];
    const float* b_dur    = (const float*)d_in[8];
    const float* w_nrg    = (const float*)d_in[9];
    const float* b_nrg    = (const float*)d_in[10];
    const float* w_pit    = (const float*)d_in[11];
    const float* b_pit    = (const float*)d_in[12];
    const float* w_proj   = (const float*)d_in[13];
    const float* b_proj   = (const float*)d_in[14];

    const int T  = in_sizes[6];            // number of frames
    const int Tp = (T + 15) & ~15;         // padded to WMMA tile rows

    float* out  = (float*)d_out;                       // x_upsamp (B,T,E)
    float* wOut = out + (size_t)BB * T * EE;           // weights  (B,L,T)

    // Workspace layout (all offsets >= 32B aligned)
    char* ws = (char*)d_ws;
    size_t off = 0;
    f16*   xcH     = (f16*)(ws + off);  off += (size_t)BB * LL * EE * sizeof(f16);   // 8 MB
    f16*   xcT     = (f16*)(ws + off);  off += (size_t)BB * EE * LL * sizeof(f16);   // 8 MB
    f16*   As      = (f16*)(ws + off);  off += (size_t)BB * Tp * LL * sizeof(f16);
    float* stds    = (float*)(ws + off); off += (size_t)BB * LL * sizeof(float);
    float* logstds = (float*)(ws + off); off += (size_t)BB * LL * sizeof(float);
    float* means   = (float*)(ws + off); off += (size_t)BB * LL * sizeof(float);
    float* denom   = (float*)(ws + off); off += (size_t)BB * T  * sizeof(float);
    (void)ws_size; (void)n_in; (void)out_size;

    // 1) xc + ranges
    k_xc_ranges<<<BB * LL, 256, 0, stream>>>(x, durf, energies, pitch, lens,
                                             w_dur, b_dur, w_nrg, b_nrg, w_pit, b_pit,
                                             w_proj, b_proj, xcH, stds, logstds);
    // 2) means
    k_means<<<BB, 32, 0, stream>>>(duri, means);
    // 3a) denominators
    k_denom<<<BB * T, 256, 0, stream>>>(frames, means, stds, logstds, lens, T, denom);
    // 3b) weights: f32 output + transposed f16 A matrix (fused)
    {
        dim3 g((T + 31) / 32, LL / 32, BB);
        k_weights_t<<<g, 256, 0, stream>>>(frames, means, stds, logstds, denom, lens,
                                           T, Tp, wOut, As);
    }
    // T1) xcH (L,E) -> xcT (E,L)
    {
        dim3 g(EE / 32, LL / 32, BB);
        k_transpose_f16<<<g, 256, 0, stream>>>(xcH, xcT);
    }
    // 4) batched WMMA GEMM: x_upsamp
    const int Mtiles = Tp / 16;
    const int totalWaves = BB * Mtiles * (EE / 64);
    const int blocks = (totalWaves + 7) / 8;           // 8 waves / 256-thread block
    k_gemm<<<blocks, 256, 0, stream>>>(As, xcT, out, T, Tp, Mtiles);
}
